// MlpBlock_4698694222120
// MI455X (gfx1250) — 8-bit matrix operand order repaired and checked by reading, compiled, not run on hardware
//
#include <hip/hip_runtime.h>
#include <hip/hip_bf16.h>

typedef __attribute__((ext_vector_type(8))) int v8i;
typedef __attribute__((ext_vector_type(4))) int v4i_t;

#define BOUND 127.0f
#define QEPS 1e-6f

#if defined(__HIP_DEVICE_COMPILE__) && __has_builtin(__builtin_amdgcn_global_load_async_to_lds_b128)
#define HAVE_ASYNC_LDS 1
#else
#define HAVE_ASYNC_LDS 0
#endif

__device__ __forceinline__ void async_copy_b128(const void* g, void* l) {
#if HAVE_ASYNC_LDS
    typedef __attribute__((address_space(1))) v4i_t* gp_t;
    typedef __attribute__((address_space(3))) v4i_t* lp_t;
    __builtin_amdgcn_global_load_async_to_lds_b128((gp_t)g, (lp_t)l, 0, 0);
#else
    *(uint4*)l = *(const uint4*)g;
#endif
}

__device__ __forceinline__ void wait_async_all() {
#if defined(__HIP_DEVICE_COMPILE__) && __has_builtin(__builtin_amdgcn_s_wait_asynccnt)
    __builtin_amdgcn_s_wait_asynccnt(0);
#endif
}

__device__ __forceinline__ int clamp_q(int q) {
    return min(127, max(-127, q));
}

__device__ __forceinline__ unsigned int pack_q4(float4 f, float s) {
    int q0 = clamp_q(__float2int_rn(f.x * s));
    int q1 = clamp_q(__float2int_rn(f.y * s));
    int q2 = clamp_q(__float2int_rn(f.z * s));
    int q3 = clamp_q(__float2int_rn(f.w * s));
    return (unsigned int)(q0 & 255) | ((unsigned int)(q1 & 255) << 8) |
           ((unsigned int)(q2 & 255) << 16) | ((unsigned int)(q3 & 255) << 24);
}

__global__ void init_max_kernel(unsigned int* maxbits) {
    maxbits[0] = 0u;
    maxbits[1] = 0u;
}

__global__ __launch_bounds__(256) void absmax_masked_kernel(
    const float* __restrict__ x, const float* __restrict__ mask,
    unsigned int* __restrict__ out_bits, long long n, int C) {
    __shared__ float red[256];
    float m = 0.0f;
    for (long long i = (long long)blockIdx.x * blockDim.x + threadIdx.x; i < n;
         i += (long long)gridDim.x * blockDim.x) {
        float v = fabsf(x[i] * mask[i / C]);
        m = fmaxf(m, v);
    }
    red[threadIdx.x] = m;
    __syncthreads();
    for (int s = 128; s > 0; s >>= 1) {
        if (threadIdx.x < s) red[threadIdx.x] = fmaxf(red[threadIdx.x], red[threadIdx.x + s]);
        __syncthreads();
    }
    if (threadIdx.x == 0) atomicMax(out_bits, __float_as_uint(red[0]));
}

__global__ __launch_bounds__(256) void quant_w_kernel(
    const float* __restrict__ W, int K, int Nout,
    signed char* __restrict__ WqT, float* __restrict__ sw) {
    __shared__ float red[256];
    const int n = blockIdx.x;
    float m = 0.0f;
    for (int k = threadIdx.x; k < K; k += 256)
        m = fmaxf(m, fabsf(W[(size_t)k * Nout + n]));
    red[threadIdx.x] = m;
    __syncthreads();
    for (int s = 128; s > 0; s >>= 1) {
        if (threadIdx.x < s) red[threadIdx.x] = fmaxf(red[threadIdx.x], red[threadIdx.x + s]);
        __syncthreads();
    }
    const float s = BOUND / fmaxf(red[0], QEPS);
    for (int k = threadIdx.x; k < K; k += 256) {
        int q = clamp_q(__float2int_rn(W[(size_t)k * Nout + n] * s));
        WqT[(size_t)n * K + k] = (signed char)q;
    }
    if (threadIdx.x == 0) sw[n] = s;
}

__global__ __launch_bounds__(256) void quant_act_kernel(
    const float* __restrict__ a, unsigned int* __restrict__ q,
    const unsigned int* __restrict__ max_bits, long long n4) {
    const float s = BOUND / fmaxf(__uint_as_float(*max_bits), QEPS);
    const float4* a4 = (const float4*)a;
    for (long long i = (long long)blockIdx.x * blockDim.x + threadIdx.x; i < n4;
         i += (long long)gridDim.x * blockDim.x) {
        q[i] = pack_q4(a4[i], s);
    }
}

__global__ __launch_bounds__(256) void gemm_aqt_i8_kernel(
    const signed char* __restrict__ Aq, const signed char* __restrict__ BqT,
    const float* __restrict__ sw, const float* __restrict__ bias,
    float* __restrict__ C, int M, int N, int K,
    const unsigned int* __restrict__ a_max_bits,
    unsigned int* out_max_bits, int relu) {
    __shared__ unsigned int lsA[2][128 * 16];
    __shared__ unsigned int lsB[2][128 * 16];

    const int tid  = threadIdx.x;
    const int lane = tid & 31;
    const int wave = tid >> 5;
    const int wm   = wave & 3;
    const int wn   = wave >> 2;
    const int half = lane >> 4;
    const int l16  = lane & 15;

    const int n0 = blockIdx.x * 128;
    const int m0 = blockIdx.y * 128;

    const float sa = BOUND / fmaxf(__uint_as_float(*a_max_bits), QEPS);

    const int lrow = tid >> 1;
    const int lseg = tid & 1;
    const signed char* gA0 = Aq + (size_t)(m0 + lrow) * K + lseg * 32;
    const signed char* gB0 = BqT + (size_t)(n0 + lrow) * K + lseg * 32;

    v8i acc[2][4] = {};

    const int ksteps = K >> 6;

    {
        unsigned int* la = &lsA[0][lrow * 16 + lseg * 8];
        unsigned int* lb = &lsB[0][lrow * 16 + lseg * 8];
        async_copy_b128(gA0, la);
        async_copy_b128(gA0 + 16, la + 4);
        async_copy_b128(gB0, lb);
        async_copy_b128(gB0 + 16, lb + 4);
    }
    wait_async_all();
    __syncthreads();

    for (int kt = 0; kt < ksteps; ++kt) {
        const int p = kt & 1;

        if (kt + 1 < ksteps) {
            const int k1 = (kt + 1) << 6;
            unsigned int* la = &lsA[p ^ 1][lrow * 16 + lseg * 8];
            unsigned int* lb = &lsB[p ^ 1][lrow * 16 + lseg * 8];
            async_copy_b128(gA0 + k1, la);
            async_copy_b128(gA0 + k1 + 16, la + 4);
            async_copy_b128(gB0 + k1, lb);
            async_copy_b128(gB0 + k1 + 16, lb + 4);
        }

        const unsigned int* bufA = lsA[p];
        const unsigned int* bufB = lsB[p];
        v8i afrag[2];
#pragma unroll
        for (int mt = 0; mt < 2; ++mt) {
            const int Mr = wm * 32 + mt * 16 + l16;
            const unsigned int base = Mr * 16 + half * 4;
            uint2 p0 = *(const uint2*)&bufA[base + 0];
            uint2 p1 = *(const uint2*)&bufA[base + 2];
            uint2 p2 = *(const uint2*)&bufA[base + 8];
            uint2 p3 = *(const uint2*)&bufA[base + 10];
            v8i a;
            a[0] = (int)p0.x; a[1] = (int)p0.y;
            a[2] = (int)p1.x; a[3] = (int)p1.y;
            a[4] = (int)p2.x; a[5] = (int)p2.y;
            a[6] = (int)p3.x; a[7] = (int)p3.y;
            afrag[mt] = a;
        }
        v8i bfrag[4];
#pragma unroll
        for (int nt = 0; nt < 4; ++nt) {
            const int Nc = wn * 64 + nt * 16 + l16;
            const unsigned int base = Nc * 16 + half * 4;
            uint4 q0 = *(const uint4*)&bufB[base + 0];
            uint4 q1 = *(const uint4*)&bufB[base + 8];
            v8i b;
            b[0] = (int)q0.x; b[1] = (int)q0.y; b[2] = (int)q0.z; b[3] = (int)q0.w;
            b[4] = (int)q1.x; b[5] = (int)q1.y; b[6] = (int)q1.z; b[7] = (int)q1.w;
            bfrag[nt] = b;
        }

#pragma unroll
        for (int mt = 0; mt < 2; ++mt)
#pragma unroll
            for (int nt = 0; nt < 4; ++nt)
                acc[mt][nt] = __builtin_amdgcn_wmma_i32_16x16x64_iu8(
                    true, afrag[mt], true, bfrag[nt], acc[mt][nt], false, false);

        if (kt + 1 < ksteps) wait_async_all();
        __syncthreads();
    }

    float lmax = 0.0f;
#pragma unroll
    for (int nt = 0; nt < 4; ++nt) {
        const int col = n0 + wn * 64 + nt * 16 + l16;
        const float inv = 1.0f / (sa * sw[col]);
        const float bv = bias[col];
#pragma unroll
        for (int mt = 0; mt < 2; ++mt) {
#pragma unroll
            for (int i = 0; i < 8; ++i) {
                const int row = m0 + wm * 32 + mt * 16 + half * 8 + i;
                float v = (float)acc[mt][nt][i] * inv + bv;
                if (relu) v = fmaxf(v, 0.0f);
                C[(size_t)row * N + col] = v;
                lmax = fmaxf(lmax, fabsf(v));
            }
        }
    }

    if (out_max_bits) {
        __syncthreads();
        float* red = (float*)&lsA[0][0];
        red[tid] = lmax;
        __syncthreads();
        for (int s = 128; s > 0; s >>= 1) {
            if (tid < s) red[tid] = fmaxf(red[tid], red[tid + s]);
            __syncthreads();
        }
        if (tid == 0) atomicMax(out_max_bits, __float_as_uint(red[0]));
    }
}

extern "C" void kernel_launch(void* const* d_in, const int* in_sizes, int n_in,
                              void* d_out, int out_size, void* d_ws, size_t ws_size,
                              hipStream_t stream) {
    const float* x    = (const float*)d_in[0];
    const float* mask = (const float*)d_in[1];
    const float* W1   = (const float*)d_in[2];
    const float* b1   = (const float*)d_in[3];
    const float* W2   = (const float*)d_in[4];
    const float* b2   = (const float*)d_in[5];
    float* out        = (float*)d_out;

    const int H = in_sizes[3];
    const int C = in_sizes[5];
    const int M = in_sizes[0] / C;

    char* ws = (char*)d_ws;
    unsigned int* maxbits = (unsigned int*)ws;
    float* sw1 = (float*)(ws + 256);
    float* sw2 = (float*)(ws + 256 + (size_t)H * 4);
    signed char* w1qT = (signed char*)(ws + 32768);
    signed char* w2qT = (signed char*)(ws + 32768 + (size_t)H * C);
    signed char* xq   = (signed char*)(ws + 32768 + 2 * (size_t)H * C);
    signed char* hq   = (signed char*)((char*)xq + (size_t)M * C);
    float* h          = (float*)((char*)hq + (size_t)M * H);

    init_max_kernel<<<1, 1, 0, stream>>>(maxbits);

    absmax_masked_kernel<<<1024, 256, 0, stream>>>(x, mask, &maxbits[0],
                                                   (long long)M * C, C);

    quant_w_kernel<<<H, 256, 0, stream>>>(W1, C, H, w1qT, sw1);
    quant_w_kernel<<<C, 256, 0, stream>>>(W2, H, C, w2qT, sw2);

    quant_act_kernel<<<2048, 256, 0, stream>>>(x, (unsigned int*)xq,
                                               &maxbits[0], (long long)M * C / 4);

    gemm_aqt_i8_kernel<<<dim3(H / 128, M / 128), 256, 0, stream>>>(
        xq, w1qT, sw1, b1, h, M, H, C, &maxbits[0], &maxbits[1], 1);

    quant_act_kernel<<<4096, 256, 0, stream>>>(h, (unsigned int*)hq,
                                               &maxbits[1], (long long)M * H / 4);

    gemm_aqt_i8_kernel<<<dim3(C / 128, M / 128), 256, 0, stream>>>(
        hq, w2qT, sw2, b2, out, M, C, H, &maxbits[1], nullptr, 0);
}
